// CharRNN_78821239816296
// MI455X (gfx1250) — matrix operand order repaired and checked by reading, not compiled, not run on hardware
//
#include <hip/hip_runtime.h>
#include <hip/hip_bf16.h>
#include <math.h>

// ---------------------------------------------------------------------------
// CharRNN (2-layer GRU, H=128) + vocab projection (V=50257) + softmax
// MI455X / gfx1250, wave32.
//   * GRU scan: exact-f32 path, V_WMMA_F32_16X16X4_F32, one persistent WG,
//     branch-free inner loops (clamped rows, K-loop split at the concat seam).
//   * Projection: TENSOR_LOAD_TO_LDS (TDM) stages the A tile; bf16
//     multiplicands / f32 accum via V_WMMA_F32_16X16X32_BF16.
//   * Softmax: online row-stats pass + probs pass with global_prefetch.
// ---------------------------------------------------------------------------

#define L_  2
#define H_  128
#define H2_ 256
#define V_  50257
#define B_  100
#define T_  50
#define NR_ (B_ * T_)   // 5000 rows of the projection

typedef __attribute__((ext_vector_type(2)))  float        v2f;
typedef __attribute__((ext_vector_type(8)))  float        v8f;
typedef __attribute__((ext_vector_type(16))) __bf16       v16bf;
typedef __attribute__((ext_vector_type(4)))  unsigned int ui32x4;
typedef __attribute__((ext_vector_type(4)))  int          i32x4;
typedef __attribute__((ext_vector_type(8)))  int          i32x8;

// ------------------------------ embedding + state init ----------------------
__global__ void embed_init_kernel(const int* __restrict__ tok,
                                  const float* __restrict__ emb,
                                  float* __restrict__ x,       // [NR_, H_]
                                  float* __restrict__ hbuf) {  // [L_, B_, H_]
    int i = blockIdx.x * 256 + threadIdx.x;            // grid sized to NR_*H_
    if (i < NR_ * H_) {
        int row = i >> 7;                              // b*T + t
        int c   = i & (H_ - 1);
        x[i] = emb[(size_t)tok[row] * H_ + c];
    }
    if (i < L_ * B_ * H_) hbuf[i] = 0.0f;
}

// ------------------------------ softmax_w -> bf16 ---------------------------
__global__ void wcvt_kernel(const float* __restrict__ w,
                            __bf16* __restrict__ wb) {
    int i = blockIdx.x * 256 + threadIdx.x;
    if (i < V_ * H_) wb[i] = (__bf16)w[i];
}

// ------------------------------ GRU scan (persistent, 1 WG, 32 waves) -------
// gate:  G = sigmoid([inp,h] @ gate_k[l] + gate_b[l])      (112x256)x(256x256)
// cand:  c = tanh   ([inp,r*h] @ cand_k[l] + cand_b[l])    (112x256)x(256x128)
// h'   = u*h + (1-u)*c
__global__ __launch_bounds__(1024) void gru_kernel(
    const float* __restrict__ x,        // [NR_, H_]  (row = b*T + t)
    const float* __restrict__ gate_k,   // [L_, 2H, 2H]
    const float* __restrict__ gate_b,   // [L_, 2H]
    const float* __restrict__ cand_k,   // [L_, 2H, H]
    const float* __restrict__ cand_b,   // [L_, H]
    float* __restrict__ hbuf,           // [L_, B_, H_]
    float* __restrict__ rh,             // [B_, H_]  scratch: r*h
    float* __restrict__ ub,             // [B_, H_]  scratch: u
    float* __restrict__ outbuf)         // [NR_, H_] layer-2 outputs
{
    const int wid  = threadIdx.x >> 5;
    const int lane = threadIdx.x & 31;
    const int hi   = lane >> 4;         // K-group select (ISA 16x4 A/B layout)
    const int lm   = lane & 15;

    for (int t = 0; t < T_; ++t) {
        for (int l = 0; l < L_; ++l) {
            const float* inp = (l == 0) ? nullptr : (hbuf + (size_t)(l - 1) * B_ * H_);
            const float* hL  = hbuf + (size_t)l * B_ * H_;

            // -------- phase 1: gate GEMM, 7x16 = 112 output tiles ----------
            {
                const float* gk = gate_k + (size_t)l * H2_ * H2_;
                for (int tile = wid; tile < 7 * 16; tile += 32) {
                    const int mt = tile >> 4, nt = tile & 15;
                    const int m0 = mt * 16, n0 = nt * 16;
                    const int m  = m0 + lm;
                    const int mc = (m < B_) ? m : (B_ - 1);   // clamp, no branch
                    const float* arow = (l == 0) ? (x + ((size_t)mc * T_ + t) * H_)
                                                 : (inp + (size_t)mc * H_);
                    const float* hrow  = hL + (size_t)mc * H_;
                    const float* gkcol = gk + n0 + lm;
                    v8f acc = {};
                    // K in [0, 128): A from input stream
#pragma unroll 4
                    for (int kk = 0; kk < H_; kk += 4) {
                        const int k0 = kk + hi * 2;
                        v2f av = *(const v2f*)(arow + k0);
                        v2f bv = {gkcol[(size_t)k0 * H2_],
                                  gkcol[(size_t)(k0 + 1) * H2_]};
                        acc = __builtin_amdgcn_wmma_f32_16x16x4_f32(
                            false, av, false, bv, (short)0, acc, false, false);
                    }
                    // K in [128, 256): A from hidden state
#pragma unroll 4
                    for (int kk = 0; kk < H_; kk += 4) {
                        const int k0 = kk + hi * 2;
                        v2f av = *(const v2f*)(hrow + k0);
                        v2f bv = {gkcol[(size_t)(H_ + k0) * H2_],
                                  gkcol[(size_t)(H_ + k0 + 1) * H2_]};
                        acc = __builtin_amdgcn_wmma_f32_16x16x4_f32(
                            false, av, false, bv, (short)0, acc, false, false);
                    }
#pragma unroll
                    for (int g = 0; g < 8; ++g) {
                        const int M = m0 + g + hi * 8;
                        const int N = n0 + lm;
                        if (M < B_) {
                            float z  = acc[g] + gate_b[l * H2_ + N];
                            float gv = 1.f / (1.f + __expf(-z));
                            if (N < H_) rh[(size_t)M * H_ + N] = gv * hL[(size_t)M * H_ + N];
                            else        ub[(size_t)M * H_ + (N - H_)] = gv;
                        }
                    }
                }
            }
            __threadfence_block();
            __syncthreads();

            // -------- phase 2: cand GEMM + state update, 7x8 = 56 tiles ----
            {
                const float* ck = cand_k + (size_t)l * H2_ * H_;
                float* hLw = hbuf + (size_t)l * B_ * H_;
                for (int tile = wid; tile < 7 * 8; tile += 32) {
                    const int mt = tile >> 3, nt = tile & 7;
                    const int m0 = mt * 16, n0 = nt * 16;
                    const int m  = m0 + lm;
                    const int mc = (m < B_) ? m : (B_ - 1);
                    const float* arow = (l == 0) ? (x + ((size_t)mc * T_ + t) * H_)
                                                 : (inp + (size_t)mc * H_);
                    const float* rrow  = rh + (size_t)mc * H_;
                    const float* ckcol = ck + n0 + lm;
                    v8f acc = {};
#pragma unroll 4
                    for (int kk = 0; kk < H_; kk += 4) {
                        const int k0 = kk + hi * 2;
                        v2f av = *(const v2f*)(arow + k0);
                        v2f bv = {ckcol[(size_t)k0 * H_],
                                  ckcol[(size_t)(k0 + 1) * H_]};
                        acc = __builtin_amdgcn_wmma_f32_16x16x4_f32(
                            false, av, false, bv, (short)0, acc, false, false);
                    }
#pragma unroll 4
                    for (int kk = 0; kk < H_; kk += 4) {
                        const int k0 = kk + hi * 2;
                        v2f av = *(const v2f*)(rrow + k0);
                        v2f bv = {ckcol[(size_t)(H_ + k0) * H_],
                                  ckcol[(size_t)(H_ + k0 + 1) * H_]};
                        acc = __builtin_amdgcn_wmma_f32_16x16x4_f32(
                            false, av, false, bv, (short)0, acc, false, false);
                    }
#pragma unroll
                    for (int g = 0; g < 8; ++g) {
                        const int M = m0 + g + hi * 8;
                        const int N = n0 + lm;
                        if (M < B_) {
                            float c  = tanhf(acc[g] + cand_b[l * H_ + N]);
                            float u  = ub[(size_t)M * H_ + N];
                            float hp = hLw[(size_t)M * H_ + N];
                            float hn = u * hp + (1.f - u) * c;
                            hLw[(size_t)M * H_ + N] = hn;
                            if (l == L_ - 1) outbuf[((size_t)M * T_ + t) * H_ + N] = hn;
                        }
                    }
                }
            }
            __threadfence_block();
            __syncthreads();
        }
    }
}

// ------------------------------ projection: logits = out @ W^T + b ----------
// Block: 8 waves. A tile (16 x 128 f32) is DMA'd into LDS by the Tensor Data
// Mover (one wave programs the D#, TENSORcnt waits, barrier releases all
// waves). Each wave then owns one 16-col B tile; K=128 -> 4 bf16 WMMAs.
__global__ __launch_bounds__(256) void proj_kernel(
    const float* __restrict__ outf,      // [NR_, H_]
    const __bf16* __restrict__ Wbf,      // [V_, H_]
    const float* __restrict__ softmax_b, // [V_]
    float* __restrict__ logits)          // [NR_, V_]
{
    __shared__ float Asf[16 * H_];       // 8 KB, LDS offset 0
    const int m0     = blockIdx.x * 16;
    const int nChunk = blockIdx.y * 128;
    const int tid    = threadIdx.x;

#if defined(__gfx1250__)
    if (tid < 32) {
        // ---- Tensor DMA Descriptor (D#): 2D tile, 4-byte elements --------
        const unsigned long long ga =
            (unsigned long long)(const void*)(outf + (size_t)m0 * H_);
        const unsigned rows = (unsigned)(NR_ - m0);   // HW zero-fills rows >= this
        ui32x4 g0;
        g0.x = 1u;                                        // count=1 (valid, user)
        g0.y = 0u;                                        // lds_addr = 0 (Asf)
        g0.z = (unsigned)(ga & 0xffffffffu);              // global_addr[31:0]
        g0.w = (unsigned)((ga >> 32) & 0x01ffffffu)       // global_addr[56:32]
             | (2u << 30);                                // type=2 ("image")
        i32x8 g1;
        g1[0] = (int)(2u << 16);                          // data_size=2 -> 4B
        g1[1] = (int)(((unsigned)H_ & 0xffffu) << 16);    // tensor_dim0 lo16
        g1[2] = (int)((rows & 0xffffu) << 16);            // dim0 hi16=0 | dim1 lo16
        g1[3] = (int)(((rows >> 16) & 0xffffu)            // dim1 hi16
             | ((unsigned)H_ << 16));                     // tile_dim0 = 128
        g1[4] = 16;                                       // tile_dim1 = 16 rows
        g1[5] = H_;                                       // tensor_dim0_stride
        g1[6] = 0;
        g1[7] = 0;
        i32x4 gz = {0, 0, 0, 0};
#if __clang_major__ >= 23
        i32x8 gz8 = {0, 0, 0, 0, 0, 0, 0, 0};
        __builtin_amdgcn_tensor_load_to_lds(g0, g1, gz, gz, gz8, 0);
#else
        __builtin_amdgcn_tensor_load_to_lds(g0, g1, gz, gz, 0);
#endif
        __builtin_amdgcn_s_wait_tensorcnt(0);
    }
#else
    for (int e = tid; e < 16 * H_; e += 256) {            // host-pass fallback
        const int r = e >> 7, c = e & (H_ - 1);
        const int row = m0 + r;
        Asf[e] = (row < NR_) ? outf[(size_t)row * H_ + c] : 0.f;
    }
#endif
    __syncthreads();

    const int wid = tid >> 5, lane = tid & 31;
    const int grp = lane >> 4, lm = lane & 15;
    const int n   = nChunk + wid * 16 + lm;               // B column (row of W)
    const int nc  = (n < V_) ? n : (V_ - 1);              // clamp, no branch
    const __bf16* brow = Wbf + (size_t)nc * H_;

    v8f acc = {};
#pragma unroll
    for (int kk = 0; kk < H_; kk += 32) {
        union { v16bf v; __bf16 e[16]; } af, bf;
        // A 16x32 bf16 layout: group grp holds K runs [grp*8,+8), [16+grp*8,+8)
        const float* pa = &Asf[lm * H_ + kk + grp * 8];
#pragma unroll
        for (int i = 0; i < 8; ++i) {
            af.e[i]     = (__bf16)pa[i];
            af.e[8 + i] = (__bf16)pa[16 + i];
        }
        // B 32x16 bf16 layout: group grp holds K runs [grp*8,+8), [16+grp*8,+8), col = lm
        const __bf16* pb = brow + kk + grp * 8;
#pragma unroll
        for (int i = 0; i < 8; ++i) { bf.e[i] = pb[i]; bf.e[8 + i] = pb[16 + i]; }
        acc = __builtin_amdgcn_wmma_f32_16x16x32_bf16(
            false, af.v, false, bf.v, (short)0, acc, false, false);
    }
#pragma unroll
    for (int g = 0; g < 8; ++g) {
        const int M = m0 + g + grp * 8;
        const int N = nChunk + wid * 16 + lm;
        if (M < NR_ && N < V_)
            logits[(size_t)M * V_ + N] = acc[g] + softmax_b[N];
    }
}

// ------------------------------ softmax row stats (online max+sum) ----------
__global__ __launch_bounds__(256) void rowstats_kernel(
    const float* __restrict__ logits, float* __restrict__ rowmax,
    float* __restrict__ rowsum)
{
    const int row = blockIdx.x;
    const int tid = threadIdx.x;
    const float* lr = logits + (size_t)row * V_;
    float m = -INFINITY, s = 0.f;
    for (int j = tid; j < V_; j += 256) {
        __builtin_prefetch(lr + j + 4096, 0, 0);   // global_prefetch_b8
        const float xv = lr[j];
        const float mn = fmaxf(m, xv);
        s = s * __expf(m - mn) + __expf(xv - mn);
        m = mn;
    }
    __shared__ float sm[256], ss[256];
    sm[tid] = m; ss[tid] = s;
    __syncthreads();
    for (int off = 128; off > 0; off >>= 1) {
        if (tid < off) {
            const float m2 = sm[tid + off], s2 = ss[tid + off];
            const float mn = fmaxf(sm[tid], m2);
            ss[tid] = ss[tid] * __expf(sm[tid] - mn) + s2 * __expf(m2 - mn);
            sm[tid] = mn;
        }
        __syncthreads();
    }
    if (tid == 0) { rowmax[row] = sm[0]; rowsum[row] = ss[0]; }
}

// ------------------------------ probs ---------------------------------------
__global__ __launch_bounds__(256) void probs_kernel(
    const float* __restrict__ logits, const float* __restrict__ rowmax,
    const float* __restrict__ rowsum, float* __restrict__ probs)
{
    const int row = blockIdx.y;
    const int col = blockIdx.x * 256 + threadIdx.x;
    if (col < V_) {
        const size_t idx = (size_t)row * V_ + col;
        probs[idx] = __expf(logits[idx] - rowmax[row]) / rowsum[row];
    }
}

// ---------------------------------------------------------------------------
extern "C" void kernel_launch(void* const* d_in, const int* in_sizes, int n_in,
                              void* d_out, int out_size, void* d_ws, size_t ws_size,
                              hipStream_t stream) {
    (void)in_sizes; (void)n_in; (void)out_size; (void)ws_size;
    const int*   tok       = (const int*)  d_in[0];
    const float* embedding = (const float*)d_in[1];
    const float* gate_k    = (const float*)d_in[2];
    const float* gate_b    = (const float*)d_in[3];
    const float* cand_k    = (const float*)d_in[4];
    const float* cand_b    = (const float*)d_in[5];
    const float* softmax_w = (const float*)d_in[6];
    const float* softmax_b = (const float*)d_in[7];

    float* logits = (float*)d_out;                       // [NR_, V_]
    float* probs  = logits + (size_t)NR_ * V_;           // [NR_, V_]

    // workspace carve-up (256B aligned)
    char*  base = (char*)d_ws;
    size_t off  = 0;
    auto carve = [&](size_t bytes) {
        void* p = base + off;
        off += (bytes + 255) & ~(size_t)255;
        return p;
    };
    float*  x      = (float*) carve((size_t)NR_ * H_ * 4);       // embeddings
    float*  outbuf = (float*) carve((size_t)NR_ * H_ * 4);       // GRU L2 out
    float*  hbuf   = (float*) carve((size_t)L_ * B_ * H_ * 4);   // hidden state
    float*  rh     = (float*) carve((size_t)B_ * H_ * 4);        // r*h
    float*  ub     = (float*) carve((size_t)B_ * H_ * 4);        // u gate
    float*  rowmax = (float*) carve((size_t)NR_ * 4);
    float*  rowsum = (float*) carve((size_t)NR_ * 4);
    __bf16* Wbf    = (__bf16*)carve((size_t)V_ * H_ * 2);        // bf16 weights

    embed_init_kernel<<<(NR_ * H_ + 255) / 256, 256, 0, stream>>>(
        tok, embedding, x, hbuf);
    wcvt_kernel<<<(V_ * H_ + 255) / 256, 256, 0, stream>>>(softmax_w, Wbf);
    gru_kernel<<<1, 1024, 0, stream>>>(
        x, gate_k, gate_b, cand_k, cand_b, hbuf, rh, ub, outbuf);
    proj_kernel<<<dim3((NR_ + 15) / 16, (V_ + 127) / 128), 256, 0, stream>>>(
        outbuf, Wbf, softmax_b, logits);
    rowstats_kernel<<<NR_, 256, 0, stream>>>(logits, rowmax, rowsum);
    probs_kernel<<<dim3((V_ + 255) / 256, NR_), 256, 0, stream>>>(
        logits, rowmax, rowsum, probs);
}
